// QDense_29205777613643
// MI455X (gfx1250) — compile-verified, matrix operand order checked by reading, not run on hardware
//
#include <hip/hip_runtime.h>
#include <hip/hip_bf16.h>

// ---------------------------------------------------------------------------
// Quaternion dense layer == ONE 4096^3 GEMM:  out = x * Wbig^T + bias_big
// Wbig = 4x4 signed block matrix (Hamilton product) of the four W_q blocks.
// Split-bf16 (hi+lo, 3 WMMA products) for ~fp32 accuracy on the bf16 pipes.
//
// Fast path (ws_size >= 128MB+16KB): prepass kernels split x / build signed
// Wbig / build bias_big once; the GEMM hot loop moves pure bf16 via async
// global->LDS copies and runs only v_wmma + ds loads (no conversion VALU).
// Fallback path: single kernel converting fp32->bf16 in-loop via v_perm_b32.
// ---------------------------------------------------------------------------

typedef __bf16 bf16_t;
typedef bf16_t v16bf __attribute__((ext_vector_type(16)));
typedef float  v8f   __attribute__((ext_vector_type(8)));
typedef int    v4i   __attribute__((vector_size(4 * sizeof(int))));
typedef __attribute__((address_space(1))) v4i *gv4i_p;   // "__device__" v4i*
typedef __attribute__((address_space(3))) v4i *lv4i_p;   // "__shared__"  v4i*

#define BM    128
#define BN    128
#define BK    32
#define KDIM  4096
#define NDIM  4096
#define QD    1024
#define ROWU4 5            // uint4 per LDS row: 4 data (16 dwords) + 1 pad
#define KT    (KDIM / BK)

#if __has_builtin(__builtin_amdgcn_global_load_async_to_lds_b128)
#define USE_ASYNC 1
#else
#define USE_ASYNC 0
#endif

__device__ __forceinline__ void waitAsync() {
#if USE_ASYNC
#if __has_builtin(__builtin_amdgcn_s_wait_asynccnt)
  __builtin_amdgcn_s_wait_asynccnt(0);
#else
  asm volatile("s_wait_asynccnt 0x0" ::: "memory");
#endif
#endif
}

// Split two fp32 (bit patterns u0,u1) into packed bf16 hi / lo dwords.
// Round-half-up via +0x8000, pack with one v_perm_b32 each.
__device__ __forceinline__ void split_pack2(unsigned u0, unsigned u1,
                                            unsigned &hp, unsigned &lp) {
  unsigned r0 = u0 + 0x8000u;
  unsigned r1 = u1 + 0x8000u;
  hp = __builtin_amdgcn_perm(r1, r0, 0x07060302u);   // [r1.hi16, r0.hi16]
  float f0 = __uint_as_float(u0) - __uint_as_float(r0 & 0xFFFF0000u);
  float f1 = __uint_as_float(u1) - __uint_as_float(r1 & 0xFFFF0000u);
  unsigned l0 = __float_as_uint(f0) + 0x8000u;
  unsigned l1 = __float_as_uint(f1) + 0x8000u;
  lp = __builtin_amdgcn_perm(l1, l0, 0x07060302u);
}

union FragU { uint2 u2[4]; v16bf v; };

// A-operand (ISA 7.12.2, 16-bit A 16x32): lane<16 K {0..7,16..23},
// lane>=16 K {8..15,24..31}.  Row stored as 8 uint2 of consecutive k
// (stride 10 uint2) -> fragment uint2 idx {0,1,4,5} + laneHi*2.
__device__ __forceinline__ v16bf load_frag(const uint2 *base, int row, int hiSel) {
  FragU fu;
  const uint2 *p = base + row * (2 * ROWU4) + hiSel * 2;
  fu.u2[0] = p[0];
  fu.u2[1] = p[1];
  fu.u2[2] = p[4];
  fu.u2[3] = p[5];
  return fu.v;
}

// ---------------------------------------------------------------------------
// Prepass 1: x (fp32) -> xhi/xlo bf16 planes, row-major [4096][4096].
// ---------------------------------------------------------------------------
__global__ __launch_bounds__(256) void split_x_kernel(
    const float *__restrict__ x, uint2 *__restrict__ xhi,
    uint2 *__restrict__ xlo) {
  const size_t i = (size_t)blockIdx.x * blockDim.x + threadIdx.x; // 4M float4s
  const uint4 u = ((const uint4 *)x)[i];
  unsigned h0, l0, h1, l1;
  split_pack2(u.x, u.y, h0, l0);
  split_pack2(u.z, u.w, h1, l1);
  xhi[i] = make_uint2(h0, h1);
  xlo[i] = make_uint2(l0, l1);
}

// ---------------------------------------------------------------------------
// Prepass 2: build signed Wbig hi/lo bf16 planes, row-major [n=4096][k=4096].
// ---------------------------------------------------------------------------
__global__ __launch_bounds__(256) void split_w_kernel(
    const float *__restrict__ Wrr, const float *__restrict__ Wri,
    const float *__restrict__ Wrj, const float *__restrict__ Wrk,
    uint2 *__restrict__ whi, uint2 *__restrict__ wlo) {
  const int Qt[4][4] = {{0,1,2,3},{1,0,3,2},{2,3,0,1},{3,2,1,0}};
  const unsigned Sx[4][4] = {{0u,0x80000000u,0x80000000u,0x80000000u},
                             {0u,0u,0u,0x80000000u},
                             {0u,0x80000000u,0u,0u},
                             {0u,0u,0x80000000u,0u}};
  const float *Wq[4] = {Wrr, Wri, Wrj, Wrk};

  const size_t i    = (size_t)blockIdx.x * blockDim.x + threadIdx.x; // 4M
  const size_t flat = i * 4;
  const int n  = (int)(flat >> 12);
  const int k  = (int)(flat & 4095);
  const int co = n >> 10, ci = k >> 10;
  const float *src = Wq[Qt[co][ci]] + (size_t)(n & (QD - 1)) * QD + (k & (QD - 1));
  const unsigned sx = Sx[co][ci];
  const uint4 u = *(const uint4 *)src;
  unsigned h0, l0, h1, l1;
  split_pack2(u.x ^ sx, u.y ^ sx, h0, l0);
  split_pack2(u.z ^ sx, u.w ^ sx, h1, l1);
  whi[i] = make_uint2(h0, h1);
  wlo[i] = make_uint2(l0, l1);
}

// ---------------------------------------------------------------------------
// Prepass 3: bias_big[n] = sum_c' S[c][c'] * b_{Q[c][c']}[o]
// ---------------------------------------------------------------------------
__global__ __launch_bounds__(256) void build_bias_kernel(
    const float *__restrict__ brr, const float *__restrict__ bri,
    const float *__restrict__ brj, const float *__restrict__ brk,
    float *__restrict__ bias) {
  const int Qt[4][4] = {{0,1,2,3},{1,0,3,2},{2,3,0,1},{3,2,1,0}};
  const float St[4][4] = {{1.f,-1.f,-1.f,-1.f},{1.f,1.f,1.f,-1.f},
                          {1.f,-1.f,1.f,1.f},{1.f,1.f,-1.f,1.f}};
  const float *Bq[4] = {brr, bri, brj, brk};
  const int n = blockIdx.x * blockDim.x + threadIdx.x; // 4096
  const int c = n >> 10, o = n & (QD - 1);
  bias[n] = St[c][0] * Bq[Qt[c][0]][o] + St[c][1] * Bq[Qt[c][1]][o] +
            St[c][2] * Bq[Qt[c][2]][o] + St[c][3] * Bq[Qt[c][3]][o];
}

// ---------------------------------------------------------------------------
// Fast GEMM: bf16 hi/lo planes already in global memory.
// ---------------------------------------------------------------------------
__global__ __launch_bounds__(256, 1) void qgemm_bf16(
    const char *__restrict__ xhi, const char *__restrict__ xlo,
    const char *__restrict__ whi, const char *__restrict__ wlo,
    const float *__restrict__ bias, float *__restrict__ out) {
  // [stage][0]=A_hi [1]=A_lo [2]=B_hi [3]=B_lo ; 80 KB
  __shared__ uint4 lds[2][4][BM * ROWU4];

  const int t = threadIdx.x, lane = t & 31, wid = t >> 5;
  const int wm = wid >> 2, wn = wid & 3;
  const int laneN = lane & 15, laneHi = lane >> 4;
  const int mbase = blockIdx.x * BM, nbase = blockIdx.y * BN;

  const int r  = t >> 1;  // tile row handled by this thread (0..127)
  const int hs = t & 1;   // segment pair: 0 -> {0,1}, 1 -> {2,3}

  const size_t aRowB = (size_t)(mbase + r) * KDIM * 2; // byte offset of row
  const size_t bRowB = (size_t)(nbase + r) * KDIM * 2;

  auto copyTiles = [&](int kt, int s) {
    const size_t kOffB = (size_t)kt * BK * 2;
    const char *g[4] = {xhi + aRowB + kOffB, xlo + aRowB + kOffB,
                        whi + bRowB + kOffB, wlo + bRowB + kOffB};
#pragma unroll
    for (int a = 0; a < 4; ++a) {
#pragma unroll
      for (int j = 0; j < 2; ++j) {
        const int seg = hs * 2 + j;                       // 16B each
        uint4 *dst = &lds[s][a][r * ROWU4 + seg];
#if USE_ASYNC
        __builtin_amdgcn_global_load_async_to_lds_b128(
            (gv4i_p)(void *)(g[a] + seg * 16), (lv4i_p)(void *)dst, 0, 0);
#else
        *dst = *(const uint4 *)(g[a] + seg * 16);
#endif
      }
    }
  };

  v8f acc[4][2];
#pragma unroll
  for (int i = 0; i < 4; ++i)
#pragma unroll
    for (int j = 0; j < 2; ++j)
#pragma unroll
      for (int e = 0; e < 8; ++e) acc[i][j][e] = 0.f;

  copyTiles(0, 0);
  waitAsync();
  __syncthreads();

  for (int kt = 0; kt < KT; ++kt) {
    const int cur = kt & 1;
    if (kt + 1 < KT) copyTiles(kt + 1, cur ^ 1); // overlap with compute

    const uint2 *Ah = (const uint2 *)lds[cur][0];
    const uint2 *Al = (const uint2 *)lds[cur][1];
    const uint2 *Bh = (const uint2 *)lds[cur][2];
    const uint2 *Bl = (const uint2 *)lds[cur][3];

    v16bf aH[4], aL[4], bHf[2], bLf[2];
#pragma unroll
    for (int sm = 0; sm < 4; ++sm) {
      const int row = wm * 64 + sm * 16 + laneN;
      aH[sm] = load_frag(Ah, row, laneHi);
      aL[sm] = load_frag(Al, row, laneHi);
    }
#pragma unroll
    for (int sn = 0; sn < 2; ++sn) {
      const int row = wn * 32 + sn * 16 + laneN;
      bHf[sn] = load_frag(Bh, row, laneHi);
      bLf[sn] = load_frag(Bl, row, laneHi);
    }

#pragma unroll
    for (int sm = 0; sm < 4; ++sm)
#pragma unroll
      for (int sn = 0; sn < 2; ++sn) {
        acc[sm][sn] = __builtin_amdgcn_wmma_f32_16x16x32_bf16(
            false, aH[sm], false, bHf[sn], (short)0, acc[sm][sn], false, false);
        acc[sm][sn] = __builtin_amdgcn_wmma_f32_16x16x32_bf16(
            false, aH[sm], false, bLf[sn], (short)0, acc[sm][sn], false, false);
        acc[sm][sn] = __builtin_amdgcn_wmma_f32_16x16x32_bf16(
            false, aL[sm], false, bHf[sn], (short)0, acc[sm][sn], false, false);
      }

    waitAsync();       // our async writes for next stage complete
    __syncthreads();   // everyone's writes visible / reads done
  }

#pragma unroll
  for (int sn = 0; sn < 2; ++sn) {
    const int n = nbase + wn * 32 + sn * 16 + laneN;
    const float bb = bias[n];
#pragma unroll
    for (int sm = 0; sm < 4; ++sm) {
      const int m0 = mbase + wm * 64 + sm * 16 + laneHi * 8;
#pragma unroll
      for (int rr = 0; rr < 8; ++rr)
        out[(size_t)(m0 + rr) * NDIM + n] = acc[sm][sn][rr] + bb;
    }
  }
}

// ---------------------------------------------------------------------------
// Fallback GEMM: convert fp32 -> bf16 hi/lo in-loop (v_perm_b32 packing).
// Used only when ws_size is too small for the prepass planes.
// ---------------------------------------------------------------------------
__device__ __forceinline__ void pack16(const unsigned *u, unsigned sx,
                                       uint4 &h0, uint4 &h1, uint4 &l0,
                                       uint4 &l1) {
  unsigned hh[8], ll[8];
#pragma unroll
  for (int j = 0; j < 8; ++j)
    split_pack2(u[2 * j] ^ sx, u[2 * j + 1] ^ sx, hh[j], ll[j]);
  h0 = make_uint4(hh[0], hh[1], hh[2], hh[3]);
  h1 = make_uint4(hh[4], hh[5], hh[6], hh[7]);
  l0 = make_uint4(ll[0], ll[1], ll[2], ll[3]);
  l1 = make_uint4(ll[4], ll[5], ll[6], ll[7]);
}

__global__ __launch_bounds__(256, 1) void qgemm_conv(
    const float *__restrict__ x,
    const float *__restrict__ Wrr, const float *__restrict__ brr,
    const float *__restrict__ Wri, const float *__restrict__ bri,
    const float *__restrict__ Wrj, const float *__restrict__ brj,
    const float *__restrict__ Wrk, const float *__restrict__ brk,
    float *__restrict__ out) {
  __shared__ uint4 lds[2][4][BM * ROWU4];

  const int t = threadIdx.x, lane = t & 31, wid = t >> 5;
  const int wm = wid >> 2, wn = wid & 3;
  const int laneN = lane & 15, laneHi = lane >> 4;
  const int mbase = blockIdx.x * BM, nbase = blockIdx.y * BN;
  const int c_out = nbase >> 10;

  const int Qt[4][4] = {{0,1,2,3},{1,0,3,2},{2,3,0,1},{3,2,1,0}};
  const float St[4][4] = {{1.f,-1.f,-1.f,-1.f},{1.f,1.f,1.f,-1.f},
                          {1.f,-1.f,1.f,1.f},{1.f,1.f,-1.f,1.f}};
  const unsigned SxT[4][4] = {{0u,0x80000000u,0x80000000u,0x80000000u},
                              {0u,0u,0u,0x80000000u},
                              {0u,0x80000000u,0u,0u},
                              {0u,0u,0x80000000u,0u}};
  const float *Wq[4] = {Wrr, Wri, Wrj, Wrk};
  const float *Bq[4] = {brr, bri, brj, brk};

  const int r = t >> 1, h = t & 1;

  unsigned aU[16], wU[16];
  unsigned curSx = 0u;

  auto loadG = [&](int kt) {
    const int k0 = kt * BK;
    const uint4 *xa =
        (const uint4 *)(x + (size_t)(mbase + r) * KDIM + k0 + h * 16);
#pragma unroll
    for (int j = 0; j < 4; ++j) {
      uint4 v = xa[j];
      aU[4 * j + 0] = v.x; aU[4 * j + 1] = v.y;
      aU[4 * j + 2] = v.z; aU[4 * j + 3] = v.w;
    }
    const int c_in = k0 >> 10;
    curSx = SxT[c_out][c_in];
    const uint4 *wa = (const uint4 *)(Wq[Qt[c_out][c_in]] +
        (size_t)((nbase & (QD - 1)) + r) * QD + (k0 & (QD - 1)) + h * 16);
#pragma unroll
    for (int j = 0; j < 4; ++j) {
      uint4 v = wa[j];
      wU[4 * j + 0] = v.x; wU[4 * j + 1] = v.y;
      wU[4 * j + 2] = v.z; wU[4 * j + 3] = v.w;
    }
  };

  auto storeL = [&](int s) {
    const int off = r * ROWU4 + h * 2;
    uint4 h0, h1, l0, l1;
    pack16(aU, 0u, h0, h1, l0, l1);
    lds[s][0][off] = h0; lds[s][0][off + 1] = h1;
    lds[s][1][off] = l0; lds[s][1][off + 1] = l1;
    pack16(wU, curSx, h0, h1, l0, l1);
    lds[s][2][off] = h0; lds[s][2][off + 1] = h1;
    lds[s][3][off] = l0; lds[s][3][off + 1] = l1;
  };

  v8f acc[4][2];
#pragma unroll
  for (int i = 0; i < 4; ++i)
#pragma unroll
    for (int j = 0; j < 2; ++j)
#pragma unroll
      for (int e = 0; e < 8; ++e) acc[i][j][e] = 0.f;

  loadG(0);
  storeL(0);
  __syncthreads();

  for (int kt = 0; kt < KT; ++kt) {
    const int cur = kt & 1;
    const bool more = (kt + 1) < KT;
    if (more) loadG(kt + 1);

    const uint2 *Ah = (const uint2 *)lds[cur][0];
    const uint2 *Al = (const uint2 *)lds[cur][1];
    const uint2 *Bh = (const uint2 *)lds[cur][2];
    const uint2 *Bl = (const uint2 *)lds[cur][3];

    v16bf aH[4], aL[4], bHf[2], bLf[2];
#pragma unroll
    for (int sm = 0; sm < 4; ++sm) {
      const int row = wm * 64 + sm * 16 + laneN;
      aH[sm] = load_frag(Ah, row, laneHi);
      aL[sm] = load_frag(Al, row, laneHi);
    }
#pragma unroll
    for (int sn = 0; sn < 2; ++sn) {
      const int row = wn * 32 + sn * 16 + laneN;
      bHf[sn] = load_frag(Bh, row, laneHi);
      bLf[sn] = load_frag(Bl, row, laneHi);
    }

#pragma unroll
    for (int sm = 0; sm < 4; ++sm)
#pragma unroll
      for (int sn = 0; sn < 2; ++sn) {
        acc[sm][sn] = __builtin_amdgcn_wmma_f32_16x16x32_bf16(
            false, aH[sm], false, bHf[sn], (short)0, acc[sm][sn], false, false);
        acc[sm][sn] = __builtin_amdgcn_wmma_f32_16x16x32_bf16(
            false, aH[sm], false, bLf[sn], (short)0, acc[sm][sn], false, false);
        acc[sm][sn] = __builtin_amdgcn_wmma_f32_16x16x32_bf16(
            false, aL[sm], false, bHf[sn], (short)0, acc[sm][sn], false, false);
      }

    if (more) {
      storeL(cur ^ 1);
      __syncthreads();
    }
  }

#pragma unroll
  for (int sn = 0; sn < 2; ++sn) {
    const int n = nbase + wn * 32 + sn * 16 + laneN;
    const int o = n & (QD - 1);
    const float bb = St[c_out][0] * Bq[Qt[c_out][0]][o] +
                     St[c_out][1] * Bq[Qt[c_out][1]][o] +
                     St[c_out][2] * Bq[Qt[c_out][2]][o] +
                     St[c_out][3] * Bq[Qt[c_out][3]][o];
#pragma unroll
    for (int sm = 0; sm < 4; ++sm) {
      const int m0 = mbase + wm * 64 + sm * 16 + laneHi * 8;
#pragma unroll
      for (int rr = 0; rr < 8; ++rr)
        out[(size_t)(m0 + rr) * NDIM + n] = acc[sm][sn][rr] + bb;
    }
  }
}

// ---------------------------------------------------------------------------
extern "C" void kernel_launch(void *const *d_in, const int *in_sizes, int n_in,
                              void *d_out, int out_size, void *d_ws,
                              size_t ws_size, hipStream_t stream) {
  // setup_inputs() dict order: x, W_rr, b_rr, W_ri, b_ri, W_rj, b_rj, W_rk, b_rk
  const float *x   = (const float *)d_in[0];
  const float *Wrr = (const float *)d_in[1];
  const float *brr = (const float *)d_in[2];
  const float *Wri = (const float *)d_in[3];
  const float *bri = (const float *)d_in[4];
  const float *Wrj = (const float *)d_in[5];
  const float *brj = (const float *)d_in[6];
  const float *Wrk = (const float *)d_in[7];
  const float *brk = (const float *)d_in[8];
  float *out = (float *)d_out;

  const size_t PLANE = (size_t)4096 * 4096 * 2;          // 32 MB bf16 plane
  const size_t NEED  = 4 * PLANE + 4096 * sizeof(float); // 128 MB + 16 KB

  dim3 grid(4096 / BM, NDIM / BN);
  dim3 block(256);

  if (ws_size >= NEED) {
    char *ws = (char *)d_ws;
    uint2 *xhi = (uint2 *)(ws + 0 * PLANE);
    uint2 *xlo = (uint2 *)(ws + 1 * PLANE);
    uint2 *whi = (uint2 *)(ws + 2 * PLANE);
    uint2 *wlo = (uint2 *)(ws + 3 * PLANE);
    float *bias = (float *)(ws + 4 * PLANE);

    split_x_kernel<<<16384, 256, 0, stream>>>(x, xhi, xlo);
    split_w_kernel<<<16384, 256, 0, stream>>>(Wrr, Wri, Wrj, Wrk, whi, wlo);
    build_bias_kernel<<<16, 256, 0, stream>>>(brr, bri, brj, brk, bias);
    qgemm_bf16<<<grid, block, 0, stream>>>((const char *)xhi, (const char *)xlo,
                                           (const char *)whi, (const char *)wlo,
                                           bias, out);
  } else {
    qgemm_conv<<<grid, block, 0, stream>>>(x, Wrr, brr, Wri, bri, Wrj, brj,
                                           Wrk, brk, out);
  }
}
